// MHSA_67302137528960
// MI455X (gfx1250) — matrix operand order repaired and checked by reading, not compiled, not run on hardware
//
#include <hip/hip_runtime.h>
#include <hip/hip_bf16.h>

#define B_  8
#define S_  1024
#define D_  768
#define H_  12
#define DH_ 64
#define LSTR 72   // padded LDS row stride (bf16 elems) to spread banks
#define NKB (S_ / 64)

typedef __attribute__((ext_vector_type(16))) __bf16 bf16x16;
typedef __attribute__((ext_vector_type(8)))  __bf16 bf16x8;
typedef __attribute__((ext_vector_type(8)))  float  f32x8;

static __device__ __forceinline__ bf16x16 cat16(bf16x8 lo, bf16x8 hi) {
    bf16x16 r;
#pragma unroll
    for (int i = 0; i < 8; ++i) { r[i] = lo[i]; r[8 + i] = hi[i]; }
    return r;
}

// ---------------------------------------------------------------------------
// Pass 1: per-head QKV projection.  One wave computes a 16-row s-tile:
//   Q/K/V[16x64] = X[16x64] * W[64x64] + b   (bf16 in, f32 WMMA accum, bf16 out)
// Block = 256 threads = 8 waves = 8 consecutive s-tiles of the SAME (b,h),
// so W^T (q,k,v) is staged once per block into LDS.
// ---------------------------------------------------------------------------
__global__ __launch_bounds__(256) void mhsa_proj_qkv(
    const float* __restrict__ X,
    const float* __restrict__ Wq, const float* __restrict__ bq,
    const float* __restrict__ Wk, const float* __restrict__ bk,
    const float* __restrict__ Wv, const float* __restrict__ bv,
    __bf16* __restrict__ Qo, __bf16* __restrict__ Ko, __bf16* __restrict__ Vo)
{
    __shared__ __attribute__((aligned(16))) __bf16 Wt[3 * DH_ * LSTR]; // [mat][dout][din]

    const int tid  = threadIdx.x;
    const int wave = tid >> 5;
    const int lane = tid & 31;
    const int n16  = lane & 15;
    const int hh   = lane >> 4;

    const int t0  = blockIdx.x * 8;                  // first s-tile of this block
    const int b   = t0 / (H_ * (S_ / 16));
    const int rem = t0 % (H_ * (S_ / 16));
    const int h   = rem / (S_ / 16);
    const int s0  = ((rem % (S_ / 16)) + wave) * 16; // wave's 16-row tile

    // --- stage W transposed (dout-major) as bf16 into LDS ---
    {
        const float* Ws[3] = {Wq, Wk, Wv};
        const int din = tid >> 2;            // 0..63
        const int do0 = (tid & 3) * 16;      // 0,16,32,48
#pragma unroll
        for (int m = 0; m < 3; ++m) {
            const float* src = Ws[m] + (size_t)h * DH_ * DH_ + (size_t)din * DH_ + do0;
#pragma unroll
            for (int j = 0; j < 16; ++j)
                Wt[m * DH_ * LSTR + (do0 + j) * LSTR + din] = (__bf16)src[j];
        }
    }
    __syncthreads();

    // --- A fragments of X (rows s0..s0+15, head slice h*64..h*64+63) ---
    bf16x16 ax[2];
    {
        const float* xrow = X + ((size_t)b * S_ + s0 + n16) * D_ + h * DH_;
#pragma unroll
        for (int f = 0; f < 2; ++f) {
            const int k0 = f * 32;
#pragma unroll
            for (int i = 0; i < 8; ++i)
                ax[f][i] = (__bf16)xrow[k0 + 8 * hh + i];
#pragma unroll
            for (int i = 0; i < 8; ++i)
                ax[f][8 + i] = (__bf16)xrow[k0 + 16 + 8 * hh + i];
        }
    }

    const float* bias3[3] = {bq, bk, bv};
    __bf16*      out3[3]  = {Qo, Ko, Vo};
#pragma unroll
    for (int m = 0; m < 3; ++m) {
        __bf16* obase = out3[m] + ((size_t)b * H_ + h) * S_ * DH_;
#pragma unroll
        for (int c = 0; c < 4; ++c) {
            const int n0 = c * 16;
            f32x8 acc = {};
#pragma unroll
            for (int f = 0; f < 2; ++f) {
                const __bf16* wp = &Wt[m * DH_ * LSTR + (n0 + n16) * LSTR + f * 32 + 8 * hh];
                bf16x16 bw = cat16(*(const bf16x8*)wp, *(const bf16x8*)(wp + 16));
                acc = __builtin_amdgcn_wmma_f32_16x16x32_bf16(
                        false, ax[f], false, bw, (short)0, acc, false, false);
            }
            const float bv_ = bias3[m][h * DH_ + n0 + n16];
#pragma unroll
            for (int i = 0; i < 8; ++i) {
                const int srow = s0 + i + 8 * hh;
                obase[(size_t)srow * DH_ + n0 + n16] = (__bf16)(acc[i] + bv_);
            }
        }
    }
}

// ---------------------------------------------------------------------------
// Pass 2: flash-attention with double-buffered async K/V staging.
// Grid = B*H*(S/128) workgroups of 256 threads; each wave owns 16 q rows.
// K and raw V are fetched with GLOBAL_LOAD_ASYNC_TO_LDS_B128 (ASYNCcnt),
// ping-pong buffered so HBM latency overlaps the WMMA + softmax compute.
// V is transposed LDS->LDS so the P*V B-fragments are contiguous loads.
// ---------------------------------------------------------------------------
__global__ __launch_bounds__(256) void mhsa_flash_attn(
    const __bf16* __restrict__ Q, const __bf16* __restrict__ K,
    const __bf16* __restrict__ V, float* __restrict__ Out)
{
    __shared__ __attribute__((aligned(16))) __bf16 Klds[2][64 * LSTR];   // [buf][key][dh] padded
    __shared__ __attribute__((aligned(16))) __bf16 Vraw[2][64 * DH_];    // [buf][key][dh] packed
    __shared__ __attribute__((aligned(16))) __bf16 Vt[64 * LSTR];        // [dh][key]
    __shared__ __attribute__((aligned(16))) __bf16 Plds[8 * 16 * LSTR];  // per-wave [row][key]

    const int tid  = threadIdx.x;
    const int wave = tid >> 5;
    const int lane = tid & 31;
    const int n16  = lane & 15;
    const int hh   = lane >> 4;

    const int nqb = S_ / 128;
    const int b   = blockIdx.x / (H_ * nqb);
    const int rem = blockIdx.x % (H_ * nqb);
    const int h   = rem / nqb;
    const int q0  = (rem % nqb) * 128 + wave * 16;

    const size_t bh  = ((size_t)b * H_ + h) * S_ * DH_;
    const __bf16* Qbh = Q + bh;
    const unsigned long long kaddr = (unsigned long long)(K + bh);
    const unsigned long long vaddr = (unsigned long long)(V + bh);

    // per-thread staging coordinates: 16B x2 per matrix per block
    const int skey = tid >> 2;             // 0..63
    const int sdh  = (tid & 3) * 16;       // 0,16,32,48

    // Q fragments (held in registers for the whole kernel)
    bf16x16 aq[2];
    {
        const __bf16* qrow = Qbh + (size_t)(q0 + n16) * DH_;
#pragma unroll
        for (int f = 0; f < 2; ++f) {
            const int k0 = f * 32;
            bf16x8 lo = *(const bf16x8*)&qrow[k0 + 8 * hh];
            bf16x8 hi = *(const bf16x8*)&qrow[k0 + 16 + 8 * hh];
            aq[f] = cat16(lo, hi);
        }
    }

    f32x8 oacc[4] = {{}, {}, {}, {}};
    float mrow[8], lrow[8];
#pragma unroll
    for (int i = 0; i < 8; ++i) { mrow[i] = -1e30f; lrow[i] = 0.f; }

    const float scale = 0.125f; // 1/sqrt(64)

    // issue one 64-key block's K + V into LDS buffer `buf` (4 async ops/wave)
    auto issue_async = [&](int buf, int kb2) {
        const unsigned k_lds = (unsigned)(unsigned long long)&Klds[buf][skey * LSTR + sdh];
        const unsigned v_lds = (unsigned)(unsigned long long)&Vraw[buf][skey * DH_ + sdh];
        const unsigned goff  = (unsigned)(((kb2 * 64 + skey) * DH_ + sdh) * 2);
        asm volatile("global_load_async_to_lds_b128 %0, %1, %2 offset:0"
                     :: "v"(k_lds), "v"(goff), "s"(kaddr) : "memory");
        asm volatile("global_load_async_to_lds_b128 %0, %1, %2 offset:16"
                     :: "v"(k_lds), "v"(goff), "s"(kaddr) : "memory");
        asm volatile("global_load_async_to_lds_b128 %0, %1, %2 offset:0"
                     :: "v"(v_lds), "v"(goff), "s"(vaddr) : "memory");
        asm volatile("global_load_async_to_lds_b128 %0, %1, %2 offset:16"
                     :: "v"(v_lds), "v"(goff), "s"(vaddr) : "memory");
    };

    issue_async(0, 0); // prologue

    for (int kb = 0; kb < NKB; ++kb) {
        const int cur = kb & 1;
        if (kb + 1 < NKB) {
            issue_async(cur ^ 1, kb + 1);
            // two batches outstanding (in-order): wait until only `next` remains
            asm volatile("s_wait_asynccnt 0x4" ::: "memory");
        } else {
            asm volatile("s_wait_asynccnt 0x0" ::: "memory");
        }
        __syncthreads();  // K[cur], Vraw[cur] resident & visible to all waves

        // ---- LDS->LDS transpose of V block: Vraw[cur] -> Vt ----
        {
            const __bf16* vr = &Vraw[cur][skey * DH_ + sdh];
            bf16x8 vlo = *(const bf16x8*)vr;
            bf16x8 vhi = *(const bf16x8*)(vr + 8);
#pragma unroll
            for (int j = 0; j < 8; ++j) Vt[(sdh + j) * LSTR + skey]     = vlo[j];
#pragma unroll
            for (int j = 0; j < 8; ++j) Vt[(sdh + 8 + j) * LSTR + skey] = vhi[j];
        }

        // ---- scores: 4 tiles of 16 keys, DH=64 contracted via 2 WMMAs ----
        f32x8 sblk[4];
#pragma unroll
        for (int j = 0; j < 4; ++j) {
            f32x8 acc = {};
#pragma unroll
            for (int f = 0; f < 2; ++f) {
                const __bf16* kp = &Klds[cur][(j * 16 + n16) * LSTR + f * 32 + 8 * hh];
                bf16x16 bk_ = cat16(*(const bf16x8*)kp, *(const bf16x8*)(kp + 16));
                acc = __builtin_amdgcn_wmma_f32_16x16x32_bf16(
                        false, aq[f], false, bk_, (short)0, acc, false, false);
            }
#pragma unroll
            for (int i = 0; i < 8; ++i) acc[i] *= scale;
            sblk[j] = acc;
        }

        // ---- online softmax: rows live across the 16 lanes of each half ----
        __bf16* pw = &Plds[wave * 16 * LSTR];
#pragma unroll
        for (int i = 0; i < 8; ++i) {
            float mx = fmaxf(fmaxf(sblk[0][i], sblk[1][i]),
                             fmaxf(sblk[2][i], sblk[3][i]));
#pragma unroll
            for (int off = 8; off >= 1; off >>= 1)
                mx = fmaxf(mx, __shfl_xor(mx, off, 32));
            const float mnew  = fmaxf(mrow[i], mx);
            const float alpha = __expf(mrow[i] - mnew);
            float rs = 0.f;
#pragma unroll
            for (int j = 0; j < 4; ++j) {
                const float p = __expf(sblk[j][i] - mnew);
                rs += p;
                pw[(i + 8 * hh) * LSTR + j * 16 + n16] = (__bf16)p;
            }
#pragma unroll
            for (int off = 8; off >= 1; off >>= 1)
                rs += __shfl_xor(rs, off, 32);
            lrow[i] = lrow[i] * alpha + rs;
            mrow[i] = mnew;
#pragma unroll
            for (int c = 0; c < 4; ++c) oacc[c][i] *= alpha;
        }

        __syncthreads();  // Vt fully built; prior-iteration readers long done

        // ---- P*V: P re-read from LDS in A layout, V^T gives contiguous B ----
#pragma unroll
        for (int f = 0; f < 2; ++f) {
            const int k0 = f * 32;
            const __bf16* prow = &Plds[wave * 16 * LSTR + n16 * LSTR];
            bf16x16 ap = cat16(*(const bf16x8*)&prow[k0 + 8 * hh],
                               *(const bf16x8*)&prow[k0 + 16 + 8 * hh]);
#pragma unroll
            for (int c = 0; c < 4; ++c) {
                const __bf16* vp = &Vt[(c * 16 + n16) * LSTR + k0 + 8 * hh];
                bf16x16 bv_ = cat16(*(const bf16x8*)vp, *(const bf16x8*)(vp + 16));
                oacc[c] = __builtin_amdgcn_wmma_f32_16x16x32_bf16(
                            false, ap, false, bv_, (short)0, oacc[c], false, false);
            }
        }
    }

    // ---- normalize and write fp32 output [B,S,D] ----
#pragma unroll
    for (int c = 0; c < 4; ++c) {
#pragma unroll
        for (int i = 0; i < 8; ++i) {
            const int srow = q0 + i + 8 * hh;
            Out[((size_t)b * S_ + srow) * D_ + h * DH_ + c * 16 + n16] =
                oacc[c][i] / lrow[i];
        }
    }
}

extern "C" void kernel_launch(void* const* d_in, const int* in_sizes, int n_in,
                              void* d_out, int out_size, void* d_ws, size_t ws_size,
                              hipStream_t stream) {
    (void)in_sizes; (void)n_in; (void)out_size; (void)ws_size;
    const float* X  = (const float*)d_in[0];
    const float* Wq = (const float*)d_in[1];
    const float* bq = (const float*)d_in[2];
    const float* Wk = (const float*)d_in[3];
    const float* bk = (const float*)d_in[4];
    const float* Wv = (const float*)d_in[5];
    const float* bv = (const float*)d_in[6];

    const size_t nqkv = (size_t)B_ * H_ * S_ * DH_;
    __bf16* Qb = (__bf16*)d_ws;
    __bf16* Kb = Qb + nqkv;
    __bf16* Vb = Kb + nqkv;
    float*  Out = (float*)d_out;

    // 6144 s-tiles / 8 waves per block = 768 blocks
    mhsa_proj_qkv<<<dim3(B_ * H_ * (S_ / 16) / 8), 256, 0, stream>>>(
        X, Wq, bq, Wk, bk, Wv, bv, Qb, Kb, Vb);
    // B*H*(S/128) = 768 blocks
    mhsa_flash_attn<<<dim3(B_ * H_ * (S_ / 128)), 256, 0, stream>>>(
        Qb, Kb, Vb, Out);
}
